// MultiheadAttention_18915035972253
// MI455X (gfx1250) — matrix operand order repaired and checked by reading, not compiled, not run on hardware
//
#include <hip/hip_runtime.h>

// ---------------------------------------------------------------------------
// MultiheadAttention forward for MI455X (gfx1250), wave32 + WMMA bf16.
// L=2048, N=4, E=2048, H=16, D=128.
// Pipeline: cvt(weights f32->bf16) -> QKV GEMMs -> flash attention -> out GEMM
// GEMM: 128x128 block tile, B staged through double-buffered LDS (8x reuse).
// ---------------------------------------------------------------------------

typedef __attribute__((ext_vector_type(16))) __bf16 v16bf;
typedef __attribute__((ext_vector_type(8)))  __bf16 v8bf;
typedef __attribute__((ext_vector_type(8)))  float  v8f;

static constexpr int   L_ = 2048;
static constexpr int   N_ = 4;
static constexpr int   E_ = 2048;
static constexpr int   H_ = 16;
static constexpr int   D_ = 128;                  // E_/H_
static constexpr float SCALE_ = 0.08838834764831845f; // D^-0.5
static constexpr float NEGF_  = -10000.0f;

static constexpr int BROW = 40;   // padded LDS row stride (elems): 20-bank step

union AFrag {
    v16bf v;
    struct { v8bf lo, hi; } p;
};

__device__ __forceinline__ v8f wmma_bf16(v16bf a, v16bf b, v8f c) {
    // D(f32 16x16) = A(16x32 bf16) * B(32x16 bf16) + C
    return __builtin_amdgcn_wmma_f32_16x16x32_bf16(
        /*neg_a=*/false, a, /*neg_b=*/false, b,
        /*c_mod=*/(short)0, c, /*reuse_a=*/false, /*reuse_b=*/false);
}

// ---------------------------------------------------------------------------
// Elementwise f32 -> bf16 (8 elems/thread, packed cvt).
// ---------------------------------------------------------------------------
__global__ void __launch_bounds__(256)
cvt_f32_bf16(const float* __restrict__ in, __bf16* __restrict__ out, int n)
{
    const int i = (blockIdx.x * blockDim.x + threadIdx.x) * 8;
    if (i < n) {
        v8f x = *(const v8f*)(in + i);
        *(v8bf*)(out + i) = __builtin_convertvector(x, v8bf);
    }
}

// ---------------------------------------------------------------------------
// GEMM:  C[M x Nout] = A[M x K] @ W[Nout x K]^T + bias
// Block = 8 waves = 128(M) x 128(N) tile; all waves share one B tile per
// K-step, staged in double-buffered LDS. A fragments loaded from global
// (f32 converted in-register when ABF16==false, bf16 direct otherwise).
// MODE 0: store f32 row-major (out projection -> d_out)
// MODE 1: store bf16 per-head (n,h,l,d)   (Q with scale=SCALE, K with scale=1)
// MODE 2: store bf16 per-head (n,h,d,l)   (V transposed)
// ---------------------------------------------------------------------------
template<int MODE, bool ABF16>
__global__ void __launch_bounds__(256)
gemm_wmma(const void* __restrict__ Aptr, const __bf16* __restrict__ W,
          const float* __restrict__ bias, float* __restrict__ outF,
          __bf16* __restrict__ outB, int M, int K, int Nout, float scale)
{
    __shared__ __align__(32) __bf16 bsh[2][128 * BROW];   // 2 x 10 KB

    const int tid    = threadIdx.x;
    const int lane   = tid & 31;
    const int wlocal = tid >> 5;               // 0..7
    const int numNb  = Nout >> 7;              // 128-wide N block tiles
    const int mb = blockIdx.x / numNb;
    const int nb = blockIdx.x - mb * numNb;
    const int m0 = mb * 128 + wlocal * 16;     // this wave's 16 rows
    const int n0 = nb * 128;                   // block's 128 cols
    const int lm = lane & 15;
    const int lg = lane >> 4;

    // cooperative B-stage mapping: thread -> (column cc, K-half hh)
    const int cc = tid >> 1;                   // 0..127
    const int hh = tid & 1;                    // 0..1 (16 K values each)

    const float*  Af = (const float*)Aptr;
    const __bf16* Ab = (const __bf16*)Aptr;

    v8f acc[8];
#pragma unroll
    for (int j = 0; j < 8; ++j)
#pragma unroll
        for (int r = 0; r < 8; ++r) acc[j][r] = 0.0f;

    int pb = 0;
    for (int k0 = 0; k0 < K; k0 += 32) {
        // ---- stage B chunk (128 cols x 32 K, bf16) into LDS, once per block
        {
            const __bf16* gp = W + (size_t)(n0 + cc) * K + k0 + 16 * hh;
            AFrag w;
            w.p.lo = *(const v8bf*)gp;
            w.p.hi = *(const v8bf*)(gp + 8);
            __bf16* dp = &bsh[pb][cc * BROW + 16 * hh];
            *(v8bf*)dp       = w.p.lo;
            *(v8bf*)(dp + 8) = w.p.hi;
        }

        // ---- A fragment: lane (lm,lg) holds row m0+lm,
        //      K = k0+8*lg..+7 and k0+16+8*lg..+7
        AFrag a;
        if (ABF16) {
            const __bf16* ap = Ab + (size_t)(m0 + lm) * K + k0 + 8 * lg;
            a.p.lo = *(const v8bf*)ap;
            a.p.hi = *(const v8bf*)(ap + 16);
            if (k0 + 32 < K) __builtin_prefetch(ap + 32, 0, 3);
        } else {
            const float* ap = Af + (size_t)(m0 + lm) * K + k0 + 8 * lg;
            a.p.lo = __builtin_convertvector(*(const v8f*)ap,        v8bf);
            a.p.hi = __builtin_convertvector(*(const v8f*)(ap + 16), v8bf);
            if (k0 + 32 < K) __builtin_prefetch(ap + 32, 0, 3);
        }

        __syncthreads();   // B tile visible; double buffer -> single barrier

        // ---- 8 WMMAs: B fragments read from LDS (conflict-free, 40-el rows)
#pragma unroll
        for (int j = 0; j < 8; ++j) {
            const __bf16* lp = &bsh[pb][(j * 16 + lm) * BROW + 8 * lg];
            AFrag b;
            b.p.lo = *(const v8bf*)lp;
            b.p.hi = *(const v8bf*)(lp + 16);
            acc[j] = wmma_bf16(a.v, b.v, acc[j]);
        }
        pb ^= 1;
    }

#pragma unroll
    for (int j = 0; j < 8; ++j) {
        const int   col = n0 + j * 16 + lm;
        const float bj  = bias[col];
#pragma unroll
        for (int r = 0; r < 8; ++r) {
            const int   row = m0 + 8 * lg + r;   // C layout: lanes 16-31 hold M=8..15
            const float val = (acc[j][r] + bj) * scale;
            if (MODE == 0) {
                outF[(size_t)row * Nout + col] = val;
            } else {
                const int l = row >> 2;          // row / N_
                const int n = row & 3;           // row % N_
                const int h = col >> 7;          // col / D_
                const int d = col & 127;         // col % D_
                if (MODE == 1)
                    outB[((size_t)(n * H_ + h) * L_ + l) * D_ + d] = (__bf16)val;
                else
                    outB[((size_t)(n * H_ + h) * D_ + d) * L_ + l] = (__bf16)val;
            }
        }
    }
}

// ---------------------------------------------------------------------------
// Flash-attention: one wave per 16-query-row tile per (n,h).
// Q,K: bf16 (n,h,l,d);  Vt: bf16 (n,h,d,l).  Output Att: bf16 (l,n,e).
// Keys consumed 32 at a time: 8 score WMMAs + softmax update + 8 O WMMAs.
// ---------------------------------------------------------------------------
__global__ void __launch_bounds__(128)
attn_wmma(const __bf16* __restrict__ Q, const __bf16* __restrict__ Km,
          const __bf16* __restrict__ Vt, __bf16* __restrict__ Att)
{
    __shared__ __align__(32) __bf16 lds[4][16 * 32];  // per-wave P staging tile

    const int lane   = threadIdx.x & 31;
    const int wlocal = threadIdx.x >> 5;
    const int wave   = blockIdx.x * 4 + wlocal;
    const int qt = wave & 127;            // L_/16 = 128 query tiles
    const int nh = wave >> 7;             // 0..N_*H_-1
    const int q0 = qt << 4;
    const int lm = lane & 15;
    const int lg = lane >> 4;

    const __bf16* Qb = Q  + (size_t)nh * L_ * D_;
    const __bf16* Kb = Km + (size_t)nh * L_ * D_;
    const __bf16* Vb = Vt + (size_t)nh * D_ * L_;

    // Q fragments for this query tile: 4 chunks of K=32 over D=128
    AFrag qf[4];
#pragma unroll
    for (int c = 0; c < 4; ++c) {
        const __bf16* qp = Qb + (size_t)(q0 + lm) * D_ + 32 * c + 8 * lg;
        qf[c].p.lo = *(const v8bf*)qp;
        qf[c].p.hi = *(const v8bf*)(qp + 16);
    }

    float mrow[8], lrow[8];
    v8f   o[8];
#pragma unroll
    for (int r = 0; r < 8; ++r) { mrow[r] = -1.0e30f; lrow[r] = 0.0f; }
#pragma unroll
    for (int t = 0; t < 8; ++t)
#pragma unroll
        for (int r = 0; r < 8; ++r) o[t][r] = 0.0f;

    for (int k0 = 0; k0 < q0 + 16; k0 += 32) {
        // ---- scores: S0 = Q * K[k0..k0+15]^T, S1 = Q * K[k0+16..k0+31]^T
        v8f s0, s1;
#pragma unroll
        for (int r = 0; r < 8; ++r) { s0[r] = 0.0f; s1[r] = 0.0f; }
#pragma unroll
        for (int c = 0; c < 4; ++c) {
            AFrag b0, b1;
            b0.p.lo = *(const v8bf*)(Kb + (size_t)(k0 + lm)      * D_ + 32 * c + 8 * lg);
            b0.p.hi = *(const v8bf*)(Kb + (size_t)(k0 + lm)      * D_ + 32 * c + 16 + 8 * lg);
            b1.p.lo = *(const v8bf*)(Kb + (size_t)(k0 + 16 + lm) * D_ + 32 * c + 8 * lg);
            b1.p.hi = *(const v8bf*)(Kb + (size_t)(k0 + 16 + lm) * D_ + 32 * c + 16 + 8 * lg);
            s0 = wmma_bf16(qf[c].v, b0.v, s0);
            s1 = wmma_bf16(qf[c].v, b1.v, s1);
        }

        // ---- causal mask (boundary chunks only)
        if (k0 + 31 > q0) {
#pragma unroll
            for (int r = 0; r < 8; ++r) {
                const int row = q0 + 8 * lg + r;
                if (k0 + lm > row)      s0[r] = NEGF_;
                if (k0 + 16 + lm > row) s1[r] = NEGF_;
            }
        }

        // ---- streaming softmax (row reductions across 16-lane C halves)
        v8f cm;
#pragma unroll
        for (int r = 0; r < 8; ++r) cm[r] = fmaxf(s0[r], s1[r]);
#pragma unroll
        for (int msk = 1; msk < 16; msk <<= 1)
#pragma unroll
            for (int r = 0; r < 8; ++r)
                cm[r] = fmaxf(cm[r], __shfl_xor(cm[r], msk, 32));

        float alpha[8];
#pragma unroll
        for (int r = 0; r < 8; ++r) {
            const float mn = fmaxf(mrow[r], cm[r]);
            alpha[r] = __expf(mrow[r] - mn);
            mrow[r]  = mn;
        }
        v8f p0, p1, ps;
#pragma unroll
        for (int r = 0; r < 8; ++r) {
            p0[r] = __expf(s0[r] - mrow[r]);
            p1[r] = __expf(s1[r] - mrow[r]);
            ps[r] = p0[r] + p1[r];
        }
#pragma unroll
        for (int msk = 1; msk < 16; msk <<= 1)
#pragma unroll
            for (int r = 0; r < 8; ++r)
                ps[r] += __shfl_xor(ps[r], msk, 32);
#pragma unroll
        for (int r = 0; r < 8; ++r) lrow[r] = lrow[r] * alpha[r] + ps[r];
#pragma unroll
        for (int t = 0; t < 8; ++t)
#pragma unroll
            for (int r = 0; r < 8; ++r) o[t][r] *= alpha[r];

        // ---- C-layout -> A-layout for P via per-wave LDS tile
        __bf16* sp = lds[wlocal];
#pragma unroll
        for (int r = 0; r < 8; ++r) {
            sp[(8 * lg + r) * 32 + lm]      = (__bf16)p0[r];
            sp[(8 * lg + r) * 32 + 16 + lm] = (__bf16)p1[r];
        }
        asm volatile("s_wait_dscnt 0" ::: "memory");
        AFrag pf;
        pf.p.lo = *(const v8bf*)(sp + lm * 32 + 8 * lg);
        pf.p.hi = *(const v8bf*)(sp + lm * 32 + 16 + 8 * lg);

        // ---- O += P(16x32) * V(32x16) for each 16-wide d slice
#pragma unroll
        for (int t = 0; t < 8; ++t) {
            AFrag vf;
            vf.p.lo = *(const v8bf*)(Vb + (size_t)(t * 16 + lm) * L_ + k0 + 8 * lg);
            vf.p.hi = *(const v8bf*)(Vb + (size_t)(t * 16 + lm) * L_ + k0 + 16 + 8 * lg);
            o[t] = wmma_bf16(pf.v, vf.v, o[t]);
        }
    }

    // ---- normalize and store to Att (l, n, e) bf16
    const int nb = nh >> 4;   // nh / H_
    const int h  = nh & 15;   // nh % H_
#pragma unroll
    for (int t = 0; t < 8; ++t) {
#pragma unroll
        for (int r = 0; r < 8; ++r) {
            const int row = q0 + 8 * lg + r;
            const int col = t * 16 + lm;
            Att[((size_t)row * N_ + nb) * E_ + h * D_ + col] =
                (__bf16)(o[t][r] / lrow[r]);
        }
    }
}

// ---------------------------------------------------------------------------
extern "C" void kernel_launch(void* const* d_in, const int* in_sizes, int n_in,
                              void* d_out, int out_size, void* d_ws, size_t ws_size,
                              hipStream_t stream)
{
    (void)in_sizes; (void)n_in; (void)out_size; (void)ws_size;

    const float* query = (const float*)d_in[0];
    const float* key   = (const float*)d_in[1];
    const float* wq    = (const float*)d_in[2];
    const float* wqb   = (const float*)d_in[3];
    const float* wk    = (const float*)d_in[4];
    const float* wkb   = (const float*)d_in[5];
    const float* wv    = (const float*)d_in[6];
    const float* wvb   = (const float*)d_in[7];
    const float* wo    = (const float*)d_in[8];
    const float* wob   = (const float*)d_in[9];

    char* ws = (char*)d_ws;
    const size_t szHead = (size_t)N_ * H_ * L_ * D_ * sizeof(__bf16); // 32 MB
    const size_t szW    = (size_t)E_ * E_ * sizeof(__bf16);           //  8 MB
    __bf16* Qb  = (__bf16*)(ws);
    __bf16* Kbf = (__bf16*)(ws +     szHead);
    __bf16* Vt  = (__bf16*)(ws + 2 * szHead);
    __bf16* Att = (__bf16*)(ws + 3 * szHead);                         // 32 MB
    __bf16* wqB = (__bf16*)(ws + 4 * szHead);
    __bf16* wkB = (__bf16*)(ws + 4 * szHead +     szW);
    __bf16* wvB = (__bf16*)(ws + 4 * szHead + 2 * szW);
    __bf16* woB = (__bf16*)(ws + 4 * szHead + 3 * szW);
    // total workspace: 4*32 MB + 4*8 MB = 160 MB

    const int M = L_ * N_;     // 8192
    const int K = E_;          // 2048
    const int Nout = E_;       // 2048

    // ---- pre-convert weights f32 -> bf16 (one pass, L2-resident thereafter)
    {
        const int n = E_ * E_;
        const dim3 cblk(256), cgrd((unsigned)(n / (256 * 8)));
        cvt_f32_bf16<<<cgrd, cblk, 0, stream>>>(wq, wqB, n);
        cvt_f32_bf16<<<cgrd, cblk, 0, stream>>>(wk, wkB, n);
        cvt_f32_bf16<<<cgrd, cblk, 0, stream>>>(wv, wvB, n);
        cvt_f32_bf16<<<cgrd, cblk, 0, stream>>>(wo, woB, n);
    }

    const dim3 blk(256);                                    // 8 waves / block
    const dim3 grd((unsigned)((M / 128) * (Nout / 128)));   // 64*16 = 1024

    // Projections: Q (scaled), K, V^T — bf16 WMMA with f32 accumulate.
    gemm_wmma<1, false><<<grd, blk, 0, stream>>>(query, wqB, wqb, nullptr, Qb,  M, K, Nout, SCALE_);
    gemm_wmma<1, false><<<grd, blk, 0, stream>>>(key,   wkB, wkb, nullptr, Kbf, M, K, Nout, 1.0f);
    gemm_wmma<2, false><<<grd, blk, 0, stream>>>(key,   wvB, wvb, nullptr, Vt,  M, K, Nout, 1.0f);

    // Flash attention: N_*H_*(L_/16) = 8192 waves, 4 waves / block.
    attn_wmma<<<dim3(2048), dim3(128), 0, stream>>>(Qb, Kbf, Vt, Att);

    // Output projection -> d_out (f32, (L,N,E)); A = bf16 Att.
    gemm_wmma<0, true><<<grd, blk, 0, stream>>>(Att, woB, wob, (float*)d_out, nullptr, M, K, Nout, 1.0f);
}
